// PointNet2Seg_26362509263296
// MI455X (gfx1250) — matrix operand order repaired and checked by reading, not compiled, not run on hardware
//
#include <hip/hip_runtime.h>
#include <hip/hip_bf16.h>

typedef _Float16 f16;
typedef __attribute__((ext_vector_type(16))) _Float16 v16h;
typedef __attribute__((ext_vector_type(8)))  _Float16 v8h;
typedef __attribute__((ext_vector_type(8)))  float    v8f;

#define INV_BN 0.9999950000374997f   // 1/sqrt(1+1e-5)

// ----------------------------------------------------------------------------
// Farthest point sampling: one block per batch, running min-distance in LDS,
// block-wide argmax (first-index tie-break to match jnp.argmax).
// ----------------------------------------------------------------------------
__global__ __launch_bounds__(256)
void fps_kernel(const float* __restrict__ xyz, int N, int npoint,
                int* __restrict__ out_idx, float* __restrict__ new_xyz)
{
  __shared__ float dist[8192];
  __shared__ float rv[256];
  __shared__ int   ri[256];
  __shared__ float cen[3];
  __shared__ int   s_far;
  const int b = blockIdx.x;
  const int t = threadIdx.x;
  const float* X = xyz + (size_t)b * N * 3;
  for (int i = t; i < N; i += 256) dist[i] = 1e10f;
  if (t == 0) s_far = 0;
  __syncthreads();
  for (int k = 0; k < npoint; ++k) {
    if (t == 0) {
      int far = s_far;
      out_idx[b * npoint + k] = far;
      float cx = X[far*3], cy = X[far*3+1], cz = X[far*3+2];
      cen[0] = cx; cen[1] = cy; cen[2] = cz;
      new_xyz[((size_t)b*npoint + k)*3 + 0] = cx;
      new_xyz[((size_t)b*npoint + k)*3 + 1] = cy;
      new_xyz[((size_t)b*npoint + k)*3 + 2] = cz;
    }
    __syncthreads();
    float cx = cen[0], cy = cen[1], cz = cen[2];
    float bv = -1.f; int bi = N;
    for (int i = t; i < N; i += 256) {
      float dx = X[i*3] - cx, dy = X[i*3+1] - cy, dz = X[i*3+2] - cz;
      float d = dx*dx + dy*dy + dz*dz;
      float m = fminf(dist[i], d);
      dist[i] = m;
      if (m > bv) { bv = m; bi = i; }
    }
    rv[t] = bv; ri[t] = bi;
    __syncthreads();
    for (int s = 128; s > 0; s >>= 1) {
      if (t < s) {
        float v2 = rv[t+s]; int i2 = ri[t+s];
        if (v2 > rv[t] || (v2 == rv[t] && i2 < ri[t])) { rv[t] = v2; ri[t] = i2; }
      }
      __syncthreads();
    }
    if (t == 0) s_far = ri[0];
    __syncthreads();
  }
}

// ----------------------------------------------------------------------------
// Ball query: one wave32 per (batch, query point); ballot + prefix-popcount
// emits the first `nsample` in-radius indices in ascending order, padded with
// the first hit (matches the reference's sort/where semantics).
// ----------------------------------------------------------------------------
__global__ void ball_query_kernel(const float* __restrict__ xyz,
                                  const float* __restrict__ new_xyz,
                                  int* __restrict__ out,
                                  int Bn, int N, int npoint, int nsample, float r2)
{
  int gid  = blockIdx.x * blockDim.x + threadIdx.x;
  int wave = gid >> 5, lane = gid & 31;
  if (wave >= Bn * npoint) return;   // wave-uniform
  int b = wave / npoint;
  const float* Xb = xyz + (size_t)b * N * 3;
  float qx = new_xyz[(size_t)wave*3], qy = new_xyz[(size_t)wave*3+1], qz = new_xyz[(size_t)wave*3+2];
  int* o = out + (size_t)wave * nsample;
  int cnt = 0, first = -1;
  for (int base = 0; base < N && cnt < nsample; base += 32) {
    int i = base + lane;
    bool ok = false;
    if (i < N) {
      float dx = Xb[i*3] - qx, dy = Xb[i*3+1] - qy, dz = Xb[i*3+2] - qz;
      ok = (dx*dx + dy*dy + dz*dz) <= r2;
    }
    unsigned mm = (unsigned)__ballot(ok);
    int prefix = __popc(mm & ((1u << lane) - 1u));
    if (ok && cnt + prefix < nsample) o[cnt + prefix] = i;
    if (first < 0 && mm) first = base + __ffs(mm) - 1;
    cnt += __popc(mm);
  }
  if (first < 0) first = 0;
  if (cnt > nsample) cnt = nsample;
  for (int s = cnt + lane; s < nsample; s += 32) o[s] = first;
}

// ----------------------------------------------------------------------------
// Weight pre-pack: W[Co x K] f32 -> Wh[Co x Kp] f16, zero-padded to Kp.
// ----------------------------------------------------------------------------
__global__ void convert_w_kernel(const float* __restrict__ W, f16* __restrict__ Wh,
                                 int K, int Kp, int total)
{
  int tid = blockIdx.x * blockDim.x + threadIdx.x;
  if (tid >= total) return;
  int c = tid / Kp, k = tid % Kp;
  Wh[tid] = (k < K) ? (f16)W[(size_t)c*K + k] : (f16)0.f;
}

// ----------------------------------------------------------------------------
// Grouped MLP input rows: [(xyz - center)/r | features[idx]] -> f16, row
// stride Cp (zero-padded). One thread per output element.
// ----------------------------------------------------------------------------
template<typename FT>
__global__ void group_kernel(const float* __restrict__ xyz, const float* __restrict__ new_xyz,
                             const FT* __restrict__ feats, const int* __restrict__ idx,
                             f16* __restrict__ Xg, int N, int npoint, int ns, int Cf,
                             int Cp, float inv_r)
{
  int C = 3 + Cf;
  int total = npoint * ns * Cp;
  int tid = blockIdx.x * blockDim.x + threadIdx.x;
  if (tid >= total) return;
  int r = tid / Cp, c = tid % Cp;
  float v = 0.f;
  if (c < C) {
    int j = r / ns;
    int i = idx[r];
    if (i < 0) i = 0;
    if (i >= N) i = N - 1;
    if (c < 3) v = (xyz[(size_t)i*3 + c] - new_xyz[(size_t)j*3 + c]) * inv_r;
    else       v = (float)feats[(size_t)i*Cf + (c - 3)];
  }
  Xg[(size_t)r * Cp + c] = (f16)v;
}

// ----------------------------------------------------------------------------
// WMMA GEMM (CDNA5 wave32): Y[M x Co] = relu((A.Wh^T + b)*INV_BN*g + be)
// A: f16, row stride lda (multiple of 32). Wh: f16, row stride Kp (padded).
// All fragment loads are contiguous vector loads; no masking anywhere.
// One wave per 16x16 tile, K consumed 32/wmma.
// ----------------------------------------------------------------------------
__global__ __launch_bounds__(256)
void gemm_wmma_kernel(const f16* __restrict__ A, int lda,
                      const f16* __restrict__ Wh, int Kp,
                      const float* __restrict__ bias, const float* __restrict__ gamma,
                      const float* __restrict__ beta, f16* __restrict__ Y,
                      int M, int Co, int bn_relu)
{
  const int lane = threadIdx.x & 31;
  const int wid  = (blockIdx.x * blockDim.x + threadIdx.x) >> 5;
  const int ntn  = Co >> 4;                 // Co always a multiple of 16
  const int ntm  = M >> 4;                  // M always a multiple of 16
  if (wid >= ntm * ntn) return;             // wave-uniform -> EXEC all-1 for WMMA
  const int tn = wid % ntn, tm = wid / ntn;
  const int hsel = lane >> 4;
  const int lr   = lane & 15;

  // A 16x32 f16 layout: elem e -> K = hsel*16 + e (16 contiguous)
  const f16* arow = A  + (size_t)(tm*16 + lr) * lda + hsel*16;
  // B 32x16 f16 layout: elem e -> K = hsel*16 + e (16 contiguous)
  const f16* brow = Wh + (size_t)(tn*16 + lr) * Kp  + hsel*16;

  v8f acc = {0.f,0.f,0.f,0.f,0.f,0.f,0.f,0.f};
  for (int kk = 0; kk < Kp; kk += 32) {
    v8h a0 = *reinterpret_cast<const v8h*>(arow + kk);
    v8h a1 = *reinterpret_cast<const v8h*>(arow + kk + 8);
    v16h av = __builtin_shufflevector(a0, a1, 0,1,2,3,4,5,6,7,8,9,10,11,12,13,14,15);
    v16h bv = *reinterpret_cast<const v16h*>(brow + kk);
    acc = __builtin_amdgcn_wmma_f32_16x16x32_f16(false, av, false, bv,
                                                 (short)0, acc, false, false);
  }

  const int n = tn*16 + lr;
  float bs = bias[n];
  float gv = bn_relu ? gamma[n] : 0.f;
  float be = bn_relu ? beta[n]  : 0.f;
  #pragma unroll
  for (int r = 0; r < 8; ++r) {            // C/D: VGPR r holds M = r + 8*hsel
    int mrow = tm*16 + r + 8*hsel;
    float vv = acc[r] + bs;
    if (bn_relu) vv = fmaxf(vv * INV_BN * gv + be, 0.f);
    Y[(size_t)mrow * Co + n] = (f16)vv;
  }
}

// ----------------------------------------------------------------------------
// Max-pool over sample dim: (npoint*ns, C) -> (npoint, C)
// ----------------------------------------------------------------------------
__global__ void maxpool_kernel(const f16* __restrict__ Yin, f16* __restrict__ out,
                               int npoint, int ns, int C)
{
  int tid = blockIdx.x * blockDim.x + threadIdx.x;
  if (tid >= npoint * C) return;
  int j = tid / C, c = tid % C;
  float m = -3.0e38f;
  for (int s = 0; s < ns; ++s)
    m = fmaxf(m, (float)Yin[((size_t)j*ns + s)*C + c]);
  out[(size_t)j*C + c] = (f16)m;
}

// ----------------------------------------------------------------------------
// Feature propagation: 3-NN inverse-distance interpolation + concat -> f16
// rows of stride Cp (zero-padded).
// ----------------------------------------------------------------------------
template<typename PT>
__global__ void interp_kernel(const float* __restrict__ xyz1, const float* __restrict__ xyz2,
                              const PT* __restrict__ points1, const f16* __restrict__ points2,
                              f16* __restrict__ Xo, int N1, int N2, int C1, int C2, int Cp)
{
  int n = blockIdx.x * blockDim.x + threadIdx.x;
  if (n >= N1) return;
  float px = xyz1[n*3], py = xyz1[n*3+1], pz = xyz1[n*3+2];
  float d0 = 1e30f, d1 = 1e30f, d2 = 1e30f;
  int   i0 = 0, i1 = 0, i2 = 0;
  for (int m = 0; m < N2; ++m) {
    float dx = xyz2[m*3] - px, dy = xyz2[m*3+1] - py, dz = xyz2[m*3+2] - pz;
    float d = dx*dx + dy*dy + dz*dz;
    if      (d < d0) { d2=d1;i2=i1; d1=d0;i1=i0; d0=d;i0=m; }
    else if (d < d1) { d2=d1;i2=i1; d1=d; i1=m; }
    else if (d < d2) { d2=d; i2=m; }
  }
  float w0 = 1.f/(d0+1e-8f), w1 = 1.f/(d1+1e-8f), w2 = 1.f/(d2+1e-8f);
  float ws = w0 + w1 + w2; w0 /= ws; w1 /= ws; w2 /= ws;
  f16* xr = Xo + (size_t)n * Cp;
  for (int c = 0; c < C1; ++c) xr[c] = (f16)(float)points1[(size_t)n*C1 + c];
  for (int c = 0; c < C2; ++c) {
    float v = w0*(float)points2[(size_t)i0*C2 + c]
            + w1*(float)points2[(size_t)i1*C2 + c]
            + w2*(float)points2[(size_t)i2*C2 + c];
    xr[C1 + c] = (f16)v;
  }
  for (int c = C1 + C2; c < Cp; ++c) xr[c] = (f16)0.f;
}

// ----------------------------------------------------------------------------
// Final 128->1 linear + sigmoid
// ----------------------------------------------------------------------------
__global__ void head2_kernel(const f16* __restrict__ X, const float* __restrict__ W2,
                             const float* __restrict__ b2, float* __restrict__ out, int M, int K)
{
  int n = blockIdx.x * blockDim.x + threadIdx.x;
  if (n >= M) return;
  float s = 0.f;
  for (int k = 0; k < K; ++k) s += (float)X[(size_t)n*K + k] * W2[k];
  s += b2[0];
  out[n] = 1.f / (1.f + expf(-s));
}

// ============================================================================
struct Layer { const float *W, *b, *g, *be; f16* Wh; int Kp; };

extern "C" void kernel_launch(void* const* d_in, const int* in_sizes, int n_in,
                              void* d_out, int out_size, void* d_ws, size_t ws_size,
                              hipStream_t stream)
{
  (void)n_in; (void)out_size; (void)ws_size;
  const int B = 16, N = 8192, CF = 7;
  const float* xyz      = (const float*)d_in[0];
  const float* features = (const float*)d_in[1];

  // ---- parameter mapping (insertion-order vs jax-sorted pytree) ----
  Layer sa1[3], sa2[3], sa3[3], fp3[2], fp2[2], fp1[3], head1[1];
  const float *W2 = nullptr, *b2 = nullptr;
  int pi = 2;
  auto take = [&](Layer* L, int nl) {
    for (int i = 0; i < nl; ++i) {
      L[i].W  = (const float*)d_in[pi+0];
      L[i].b  = (const float*)d_in[pi+1];
      L[i].g  = (const float*)d_in[pi+2];
      L[i].be = (const float*)d_in[pi+3];
      pi += 4;
    }
  };
  if (in_sizes[2] == 64*10) {  // insertion order: sa1,sa2,sa3,fp3,fp2,fp1,head1,head2
    take(sa1,3); take(sa2,3); take(sa3,3);
    take(fp3,2); take(fp2,2); take(fp1,3); take(head1,1);
    W2 = (const float*)d_in[pi]; b2 = (const float*)d_in[pi+1];
  } else {                     // sorted keys: fp1,fp2,fp3,head1,head2,sa1,sa2,sa3
    take(fp1,3); take(fp2,2); take(fp3,2); take(head1,1);
    W2 = (const float*)d_in[pi]; b2 = (const float*)d_in[pi+1]; pi += 2;
    take(sa1,3); take(sa2,3); take(sa3,3);
  }

  // ---- workspace carve (~21 MB total) ----
  char* wp = (char*)d_ws;
  auto carve = [&](size_t bytes) -> void* {
    void* r = (void*)wp; wp += (bytes + 255) & ~(size_t)255; return r;
  };
  float* l1_xyz  = (float*)carve((size_t)B*256*3*4);
  float* l2_xyz  = (float*)carve((size_t)B*128*3*4);
  float* l3_xyz  = (float*)carve((size_t)B* 64*3*4);
  f16*   l1_feat = (f16*)  carve((size_t)B*256*128*2);
  f16*   l2_feat = (f16*)  carve((size_t)B*128*256*2);
  f16*   l3_feat = (f16*)  carve((size_t)B* 64*512*2);
  f16*   l2_fp   = (f16*)  carve((size_t)B*128*256*2);
  f16*   l1_fp   = (f16*)  carve((size_t)B*256*128*2);
  int*   fps_idx = (int*)  carve((size_t)B*256*4);
  int*   ball_idx= (int*)  carve((size_t)B*256*32*4);
  f16*   bufA    = (f16*)  carve((size_t)8192*288*2);  // max padded X: 8192 x 288
  f16*   bufB    = (f16*)  carve((size_t)8192*512*2);  // widest layer out

  auto cdiv = [](int a, int b){ return (a + b - 1) / b; };

  // ---- pre-pack all weights to padded f16 (once per launch, on-stream) ----
  auto prep = [&](Layer& L, int K, int Co) {
    int Kp = (K + 31) & ~31;
    L.Kp = Kp;
    L.Wh = (f16*)carve((size_t)Co * Kp * 2);
    int total = Co * Kp;
    convert_w_kernel<<<cdiv(total,256), 256, 0, stream>>>(L.W, L.Wh, K, Kp, total);
  };
  prep(sa1[0], 10, 64);   prep(sa1[1], 64, 64);   prep(sa1[2], 64, 128);
  prep(sa2[0], 131, 128); prep(sa2[1], 128, 128); prep(sa2[2], 128, 256);
  prep(sa3[0], 259, 256); prep(sa3[1], 256, 256); prep(sa3[2], 256, 512);
  prep(fp3[0], 768, 256); prep(fp3[1], 256, 256);
  prep(fp2[0], 384, 256); prep(fp2[1], 256, 128);
  prep(fp1[0], 135, 128); prep(fp1[1], 128, 128); prep(fp1[2], 128, 128);
  prep(head1[0], 128, 128);

  auto gemm = [&](const f16* A, int lda, const Layer& L, f16* Y, int M, int Co) {
    int tiles  = (M/16) * (Co/16);
    int blocks = (tiles*32 + 255) / 256;
    gemm_wmma_kernel<<<blocks, 256, 0, stream>>>(A, lda, L.Wh, L.Kp,
                                                 L.b, L.g, L.be, Y, M, Co, 1);
  };

  // =========================== SA1: 8192 -> 256 ===========================
  fps_kernel<<<B, 256, 0, stream>>>(xyz, N, 256, fps_idx, l1_xyz);
  { int waves = B*256;
    ball_query_kernel<<<cdiv(waves*32,256), 256, 0, stream>>>(
        xyz, l1_xyz, ball_idx, B, N, 256, 32, 0.05f*0.05f); }
  for (int b = 0; b < B; ++b) {
    int M = 256*32, Cp = 32;   // K = 10 -> Kp 32
    group_kernel<float><<<cdiv(M*Cp,256), 256, 0, stream>>>(
        xyz + (size_t)b*N*3, l1_xyz + (size_t)b*256*3,
        features + (size_t)b*N*CF, ball_idx + (size_t)b*256*32,
        bufA, N, 256, 32, CF, Cp, 1.f/0.05f);
    gemm(bufA, 32,  sa1[0], bufB, M, 64);
    gemm(bufB, 64,  sa1[1], bufA, M, 64);
    gemm(bufA, 64,  sa1[2], bufB, M, 128);
    maxpool_kernel<<<cdiv(256*128,256), 256, 0, stream>>>(
        bufB, l1_feat + (size_t)b*256*128, 256, 32, 128);
  }

  // =========================== SA2: 256 -> 128 ============================
  fps_kernel<<<B, 256, 0, stream>>>(l1_xyz, 256, 128, fps_idx, l2_xyz);
  { int waves = B*128;
    ball_query_kernel<<<cdiv(waves*32,256), 256, 0, stream>>>(
        l1_xyz, l2_xyz, ball_idx, B, 256, 128, 64, 0.1f*0.1f); }
  for (int b = 0; b < B; ++b) {
    int M = 128*64, Cp = 160;  // K = 131 -> Kp 160
    group_kernel<f16><<<cdiv(M*Cp,256), 256, 0, stream>>>(
        l1_xyz + (size_t)b*256*3, l2_xyz + (size_t)b*128*3,
        l1_feat + (size_t)b*256*128, ball_idx + (size_t)b*128*64,
        bufA, 256, 128, 64, 128, Cp, 1.f/0.1f);
    gemm(bufA, 160, sa2[0], bufB, M, 128);
    gemm(bufB, 128, sa2[1], bufA, M, 128);
    gemm(bufA, 128, sa2[2], bufB, M, 256);
    maxpool_kernel<<<cdiv(128*256,256), 256, 0, stream>>>(
        bufB, l2_feat + (size_t)b*128*256, 128, 64, 256);
  }

  // ============================ SA3: 128 -> 64 ============================
  fps_kernel<<<B, 256, 0, stream>>>(l2_xyz, 128, 64, fps_idx, l3_xyz);
  { int waves = B*64;
    ball_query_kernel<<<cdiv(waves*32,256), 256, 0, stream>>>(
        l2_xyz, l3_xyz, ball_idx, B, 128, 64, 128, 0.2f*0.2f); }
  for (int b = 0; b < B; ++b) {
    int M = 64*128, Cp = 288;  // K = 259 -> Kp 288
    group_kernel<f16><<<cdiv(M*Cp,256), 256, 0, stream>>>(
        l2_xyz + (size_t)b*128*3, l3_xyz + (size_t)b*64*3,
        l2_feat + (size_t)b*128*256, ball_idx + (size_t)b*64*128,
        bufA, 128, 64, 128, 256, Cp, 1.f/0.2f);
    gemm(bufA, 288, sa3[0], bufB, M, 256);
    gemm(bufB, 256, sa3[1], bufA, M, 256);
    gemm(bufA, 256, sa3[2], bufB, M, 512);
    maxpool_kernel<<<cdiv(64*512,256), 256, 0, stream>>>(
        bufB, l3_feat + (size_t)b*64*512, 64, 128, 512);
  }

  // ======================= FP3: (l2 <- l3), 768->256->256 =================
  for (int b = 0; b < B; ++b) {
    interp_kernel<f16><<<cdiv(128,256), 256, 0, stream>>>(
        l2_xyz + (size_t)b*128*3, l3_xyz + (size_t)b*64*3,
        l2_feat + (size_t)b*128*256, l3_feat + (size_t)b*64*512,
        bufA, 128, 64, 256, 512, 768);
    gemm(bufA, 768, fp3[0], bufB, 128, 256);
    gemm(bufB, 256, fp3[1], l2_fp + (size_t)b*128*256, 128, 256);
  }

  // ======================= FP2: (l1 <- l2), 384->256->128 =================
  for (int b = 0; b < B; ++b) {
    interp_kernel<f16><<<cdiv(256,256), 256, 0, stream>>>(
        l1_xyz + (size_t)b*256*3, l2_xyz + (size_t)b*128*3,
        l1_feat + (size_t)b*256*128, l2_fp + (size_t)b*128*256,
        bufA, 256, 128, 128, 256, 384);
    gemm(bufA, 384, fp2[0], bufB, 256, 256);
    gemm(bufB, 256, fp2[1], l1_fp + (size_t)b*256*128, 256, 128);
  }

  // ============ FP1 + head: 135->128->128->128 ->128 ->1 ==================
  for (int b = 0; b < B; ++b) {
    interp_kernel<float><<<cdiv(N,256), 256, 0, stream>>>(
        xyz + (size_t)b*N*3, l1_xyz + (size_t)b*256*3,
        features + (size_t)b*N*CF, l1_fp + (size_t)b*256*128,
        bufA, N, 256, CF, 128, 160);              // K = 135 -> Kp 160
    gemm(bufA, 160, fp1[0], bufB, N, 128);
    gemm(bufB, 128, fp1[1], bufA, N, 128);
    gemm(bufA, 128, fp1[2], bufB, N, 128);
    gemm(bufB, 128, head1[0], bufA, N, 128);
    head2_kernel<<<cdiv(N,256), 256, 0, stream>>>(
        bufA, W2, b2, (float*)d_out + (size_t)b*N, N, 128);
  }
}
